// Encoder_32349693673701
// MI455X (gfx1250) — matrix operand order repaired and checked by reading, not compiled, not run on hardware
//
#include <hip/hip_runtime.h>

typedef __attribute__((ext_vector_type(16))) _Float16 v16h;
typedef __attribute__((ext_vector_type(8)))  _Float16 v8h;
typedef __attribute__((ext_vector_type(8)))  float    v8f;
typedef __attribute__((ext_vector_type(4)))  float    v4f;

// Problem dims
#define BB    128
#define TT    128
#define KK    512
#define UNITS 512
#define NG    2048   // 4*UNITS

static __device__ __forceinline__ v16h combine16(v8h a, v8h b) {
    return __builtin_shufflevector(a, b, 0,1,2,3,4,5,6,7,8,9,10,11,12,13,14,15);
}

// A-fragment (16x32 f16, M x K) from row-major matrix with leading dim ld.
// Layout (ISA 7.12.2): lanes 0-15 -> M=lane, halves = K{0..7, 16..23};
//                      lanes 16-31 -> M=lane-16, halves = K{8..15, 24..31}.
static __device__ __forceinline__ v16h load_a_frag(const _Float16* __restrict__ p,
                                                   int ld, int rowBase, int kBase, int lane) {
    int row  = rowBase + (lane & 15);
    int koff = (lane >> 4) * 8;
    const _Float16* q = p + (size_t)row * ld + kBase + koff;
    v8h x0 = *(const v8h*)q;
    v8h x1 = *(const v8h*)(q + 16);
    return combine16(x0, x1);
}

// B-fragment (32x16 f16, K x N) from row-major [K][N] matrix with leading dim ld.
// Layout: N = lane%16; halves = K{0..7, 16..23} (lanes 0-15), K{8..15, 24..31} (lanes 16-31).
static __device__ __forceinline__ v16h load_b_frag(const _Float16* __restrict__ p,
                                                   int ld, int kBase, int nBase, int lane) {
    const _Float16* q = p + (size_t)(kBase + (lane >> 4) * 8) * ld + nBase + (lane & 15);
    v16h r;
#pragma unroll
    for (int s = 0; s < 8; ++s) {
        r[s]     = q[(size_t)s * ld];
        r[s + 8] = q[(size_t)(s + 16) * ld];
    }
    return r;
}

// ---------------- f32 -> f16 conversion (8 elements / thread) ----------------
__global__ void __launch_bounds__(256)
cvt_f16_kernel(const float* __restrict__ src, _Float16* __restrict__ dst, int n) {
    int base = (blockIdx.x * 256 + threadIdx.x) * 8;
    if (base >= n) return;
    v4f a = *(const v4f*)(src + base);
    v4f b = *(const v4f*)(src + base + 4);
    v8h o;
    o[0]=(_Float16)a[0]; o[1]=(_Float16)a[1]; o[2]=(_Float16)a[2]; o[3]=(_Float16)a[3];
    o[4]=(_Float16)b[0]; o[5]=(_Float16)b[1]; o[6]=(_Float16)b[2]; o[7]=(_Float16)b[3];
    *(v8h*)(dst + base) = o;
}

// ---------------- Phase A: xzT[t][b][g] = (x @ W)[b*T+t][g] + bias[g] ----------------
// grid = (NG/64, (B*T)/128), block = 256 (8 waves). Wave w does rows mWave..mWave+15,
// cols nBase..nBase+63 (4 WMMA tiles), K loop over 512 by 32.
__global__ void __launch_bounds__(256)
gemm_xw_kernel(const _Float16* __restrict__ xh, const _Float16* __restrict__ Wh,
               const float* __restrict__ bias, float* __restrict__ xzT) {
    const int tid  = threadIdx.x;
    const int lane = tid & 31;
    const int w    = tid >> 5;
    const int mWave = blockIdx.y * 128 + w * 16;
    const int nBase = blockIdx.x * 64;

    v8f acc[4] = {};
    for (int kc = 0; kc < KK; kc += 32) {
        v16h a = load_a_frag(xh, KK, mWave, kc, lane);
#pragma unroll
        for (int q = 0; q < 4; ++q) {
            v16h b = load_b_frag(Wh, NG, kc, nBase + q * 16, lane);
            acc[q] = __builtin_amdgcn_wmma_f32_16x16x32_f16(
                false, a, false, b, (short)0, acc[q], false, false);
        }
    }

    const int nl = lane & 15;
    const int rh = lane >> 4;   // 0/1 -> M += 0/8
#pragma unroll
    for (int q = 0; q < 4; ++q) {
        int   ncol = nBase + q * 16 + nl;
        float bv   = bias[ncol];
#pragma unroll
        for (int r = 0; r < 8; ++r) {
            int mrow = mWave + r + rh * 8;        // global M = b*T + t
            int t    = mrow & (TT - 1);
            int bi   = mrow >> 7;
            xzT[((size_t)t * BB + bi) * NG + ncol] = acc[q][r] + bv;
        }
    }
}

// ---------------- barrier counter init ----------------
__global__ void init_cnt_kernel(unsigned* cnt) {
    __hip_atomic_store(cnt, 0u, __ATOMIC_RELAXED, __HIP_MEMORY_SCOPE_AGENT);
}

// ---------------- Phase B: persistent recurrent scan ----------------
// grid = UNITS/16 = 32 blocks, block = 256 (8 waves). Block owns units u0..u0+15,
// i.e. gate columns {q*512+u0..+15 : q=0..3}. Wave w owns batch rows 16w..16w+15.
// c-state lives in registers; h (f16) goes through global each step + grid barrier.
#define ULD 72  // padded LDS row stride in halves (bank-conflict-free for lane-indexed b128)
__global__ void __launch_bounds__(256)
lstm_scan_kernel(const _Float16* __restrict__ Uh, const float* __restrict__ xzT,
                 _Float16* __restrict__ hh, float* __restrict__ out, unsigned* cnt) {
    __shared__ _Float16 Ulds[KK * ULD];

    const int tid  = threadIdx.x;
    const int lane = tid & 31;
    const int w    = tid >> 5;
    const int u0   = blockIdx.x * 16;

    // Stage U slice [512 K][64 gate-cols] into LDS (64 KB + pad) once, using the
    // gfx1250 async memory->LDS path (no VGPR round-trip, tracked by ASYNCcnt).
    for (int idx = tid; idx < KK * 8; idx += 256) {
        int k  = idx >> 3;
        int c  = idx & 7;
        int q  = c >> 1;
        int h8 = (c & 1) * 8;
        unsigned           laddr = (unsigned)(size_t)(Ulds + k * ULD + q * 16 + h8);
        unsigned long long gaddr = (unsigned long long)(const void*)
                                   (Uh + (size_t)k * NG + q * UNITS + u0 + h8);
        asm volatile("global_load_async_to_lds_b128 %0, %1, off"
                     :: "v"(laddr), "v"(gaddr) : "memory");
    }
    asm volatile("s_wait_asynccnt 0" ::: "memory");
    __syncthreads();

    const int nl   = lane & 15;
    const int rh   = lane >> 4;
    const int rowA = w * 16 + nl;     // batch row for A-fragment
    const int unit = u0 + nl;

    float cs[8];
#pragma unroll
    for (int r = 0; r < 8; ++r) cs[r] = 0.f;

    const unsigned nwg = gridDim.x;

    for (int t = 0; t < TT; ++t) {
        v8f acc[4] = {};
        if (t > 0) {
            for (int kc = 0; kc < UNITS; kc += 32) {
                const _Float16* ap = hh + (size_t)rowA * UNITS + kc + rh * 8;
                v16h a = combine16(*(const v8h*)ap, *(const v8h*)(ap + 16));
#pragma unroll
                for (int q = 0; q < 4; ++q) {
                    const _Float16* bp = Ulds + (kc + rh * 8) * ULD + q * 16 + nl;
                    v16h b;
#pragma unroll
                    for (int s = 0; s < 8; ++s) {
                        b[s]     = bp[s * ULD];
                        b[s + 8] = bp[(s + 16) * ULD];
                    }
                    acc[q] = __builtin_amdgcn_wmma_f32_16x16x32_f16(
                        false, a, false, b, (short)0, acc[q], false, false);
                }
            }
        }
        // gates + state update for this lane's 8 (row, unit) cells
#pragma unroll
        for (int r = 0; r < 8; ++r) {
            int row = w * 16 + r + rh * 8;
            const float* z = xzT + ((size_t)t * BB + row) * NG;
            float zi = acc[0][r] + z[0 * UNITS + unit];
            float zf = acc[1][r] + z[1 * UNITS + unit];
            float zg = acc[2][r] + z[2 * UNITS + unit];
            float zo = acc[3][r] + z[3 * UNITS + unit];
            float ig = 1.f / (1.f + __expf(-zi));
            float fg = 1.f / (1.f + __expf(-zf));
            float gg = tanhf(zg);
            float og = 1.f / (1.f + __expf(-zo));
            float cn = fg * cs[r] + ig * gg;
            cs[r] = cn;
            float hn = og * tanhf(cn);
            hh[(size_t)row * UNITS + unit] = (_Float16)hn;
            if (t == TT - 1) {
                size_t oi = (size_t)row * UNITS + unit;
                out[oi]                          = hn;  // state_h copy 1
                out[(size_t)BB * UNITS + oi]     = hn;  // state_h copy 2
                out[(size_t)2 * BB * UNITS + oi] = cn;  // state_c
            }
        }
        if (t < TT - 1) {
            __threadfence();          // release h stores to device scope
            __syncthreads();
            if (tid == 0) {
                __hip_atomic_fetch_add(cnt, 1u, __ATOMIC_RELEASE, __HIP_MEMORY_SCOPE_AGENT);
                unsigned target = (unsigned)(t + 1) * nwg;
                while (__hip_atomic_load(cnt, __ATOMIC_ACQUIRE, __HIP_MEMORY_SCOPE_AGENT) < target)
                    __builtin_amdgcn_s_sleep(2);
            }
            __syncthreads();
            __threadfence();          // acquire: invalidate before reading fresh h
        }
    }
}

extern "C" void kernel_launch(void* const* d_in, const int* in_sizes, int n_in,
                              void* d_out, int out_size, void* d_ws, size_t ws_size,
                              hipStream_t stream) {
    const float* x  = (const float*)d_in[0];   // [B,T,K]
    const float* W  = (const float*)d_in[1];   // [K,4U]
    const float* U  = (const float*)d_in[2];   // [U,4U]
    const float* bb = (const float*)d_in[3];   // [4U]
    float* out = (float*)d_out;

    char* ws = (char*)d_ws;
    const size_t xh_bytes  = (size_t)BB * TT * KK * 2;       // 16 MB  f16 x
    const size_t wh_bytes  = (size_t)KK * NG * 2;            // 2 MB   f16 W
    const size_t uh_bytes  = (size_t)UNITS * NG * 2;         // 2 MB   f16 U
    const size_t xz_bytes  = (size_t)BB * TT * NG * 4;       // 134 MB f32 xz (time-major)
    const size_t hh_bytes  = (size_t)BB * UNITS * 2;         // 128 KB f16 h

    _Float16* xh  = (_Float16*)(ws);
    _Float16* Wh  = (_Float16*)(ws + xh_bytes);
    _Float16* Uh  = (_Float16*)(ws + xh_bytes + wh_bytes);
    float*    xzT = (float*)   (ws + xh_bytes + wh_bytes + uh_bytes);
    _Float16* hh  = (_Float16*)(ws + xh_bytes + wh_bytes + uh_bytes + xz_bytes);
    unsigned* cnt = (unsigned*)(ws + xh_bytes + wh_bytes + uh_bytes + xz_bytes + hh_bytes);

    // f32 -> f16 conversions (layout preserved; no transposes needed)
    {
        int n = BB * TT * KK;      // 8388608
        cvt_f16_kernel<<<n / (256 * 8), 256, 0, stream>>>(x, xh, n);
    }
    {
        int n = KK * NG;           // 1048576
        cvt_f16_kernel<<<n / (256 * 8), 256, 0, stream>>>(W, Wh, n);
    }
    {
        int n = UNITS * NG;        // 1048576
        cvt_f16_kernel<<<n / (256 * 8), 256, 0, stream>>>(U, Uh, n);
    }

    // Phase A: big WMMA GEMM, writes time-major xz with bias folded in
    gemm_xw_kernel<<<dim3(NG / 64, (BB * TT) / 128), 256, 0, stream>>>(xh, Wh, bb, xzT);

    // Phase B: persistent recurrent scan with grid barrier
    init_cnt_kernel<<<1, 1, 0, stream>>>(cnt);
    lstm_scan_kernel<<<UNITS / 16, 256, 0, stream>>>(Uh, xzT, hh, out, cnt);
}
